// EncoderGNN_65085934403996
// MI455X (gfx1250) — matrix operand order repaired and checked by reading, not compiled, not run on hardware
//
#include <hip/hip_runtime.h>

// ---------------------------------------------------------------------------
// EncoderGNN for MI455X (gfx1250): bf16 WMMA (v_wmma_f32_16x16x32_bf16),
// f32 accumulate. Edge kernel: M=64/block, LDS-staged B slices (async
// global->LDS), fused bias+ReLU+LayerNorm+layer2+scatter.
// ---------------------------------------------------------------------------

typedef __bf16 bf16;
typedef __attribute__((ext_vector_type(16))) __bf16        bf16x16;
typedef __attribute__((ext_vector_type(8)))  float         f32x8;
typedef __attribute__((ext_vector_type(4)))  unsigned int  u32x4;
typedef int gvec_i32x4 __attribute__((vector_size(16)));   // matches builtin V4i

#define NSPA   65160
#define NSPH   2883
#define NEDGE  131072
#define DD     256
#define KPAD0  544      // 3 (padded to 32) + 256 + 256
#define KPAD1  768      // 256 + 256 + 256
#define KPADN  512      // 256 + 256

// LDS strides padded so 16 rows land on distinct banks & stay 16B aligned
#define KA_E   776      // bf16, edge A rows (stride 1552B)
#define KA_N   520      // bf16, node A rows
#define HS     260      // f32  h stride
#define HBS    264      // bf16 hhat stride
#define BLS    40       // bf16, B-slice row stride (80B: 20 banks, 16B aligned)

union AF { bf16x16 v; u32x4 u[2]; };

__device__ __forceinline__ f32x8 wmma_bf16(const bf16x16& a, const bf16x16& b,
                                           const f32x8& c) {
  return __builtin_amdgcn_wmma_f32_16x16x32_bf16(false, a, false, b,
                                                 (short)0, c, false, false);
}

// ---- global -> LDS 16B copy: async-to-LDS path on gfx1250 when available ----
#if defined(__AMDGCN__) && \
    __has_builtin(__builtin_amdgcn_global_load_async_to_lds_b128) && \
    __has_builtin(__builtin_amdgcn_s_wait_asynccnt)
#define HAS_ASYNC_LDS 1
#else
#define HAS_ASYNC_LDS 0
#endif

__device__ __forceinline__ void g2l_b128(void* lds, const void* g) {
#if HAS_ASYNC_LDS
  __builtin_amdgcn_global_load_async_to_lds_b128(
      (__attribute__((address_space(1))) gvec_i32x4*)g,
      (__attribute__((address_space(3))) gvec_i32x4*)lds,
      0, 0);
#else
  *(u32x4*)lds = *(const u32x4*)g;
#endif
}
__device__ __forceinline__ void async_fence() {
#if HAS_ASYNC_LDS
  __builtin_amdgcn_s_wait_asynccnt(0);
#endif
}

// ---------------- prep kernels ----------------

__global__ void cvt_bf16_kernel(const float* __restrict__ src,
                                bf16* __restrict__ dst, int n) {
  int i = blockIdx.x * 256 + threadIdx.x;
  if (i < n) dst[i] = (bf16)src[i];
}

// Wt[n][k] = W[k][n]  (N fixed = 256), bf16
__global__ void prep_wt_kernel(const float* __restrict__ W,
                               bf16* __restrict__ Wt, int K) {
  int idx = blockIdx.x * 256 + threadIdx.x;
  if (idx >= 256 * K) return;
  int n = idx / K, k = idx % K;
  Wt[(size_t)n * K + k] = (bf16)W[(size_t)k * 256 + n];
}

// We1a (515x256) -> padded transposed bf16 [256][544]
__global__ void prep_w1a_kernel(const float* __restrict__ W,
                                bf16* __restrict__ Wt) {
  const int K = KPAD0;
  int idx = blockIdx.x * 256 + threadIdx.x;
  if (idx >= 256 * K) return;
  int n = idx / K, k = idx % K;
  int src = (k < 3) ? k : (k >= 32 ? k - 29 : -1);
  float v = (src >= 0) ? W[(size_t)src * 256 + n] : 0.0f;
  Wt[(size_t)n * K + k] = (bf16)v;
}

__global__ void zero_f32_kernel(float* __restrict__ p, int n) {
  int i = blockIdx.x * 256 + threadIdx.x;
  if (i < n) p[i] = 0.0f;
}

// ---------------- fused edge MLP + scatter ----------------
// grid = NEDGE/64 blocks, 256 threads (8 waves). Each block: 64 edges x 256.
// Wave w: row band = w>>1 (16 rows), col half = w&1 (128 cols) -> 8 tiles.
__global__ __launch_bounds__(256)
void edge_mlp_kernel(const float* __restrict__ edges0,      // step0 raw (Nx3)
                     const bf16*  __restrict__ edges_prev,  // steps>0 (NxD)
                     const bf16*  __restrict__ nodes_bf,
                     const int*   __restrict__ senders,
                     const int*   __restrict__ receivers,
                     const bf16*  __restrict__ W1t, int kpad,
                     const float* __restrict__ be1,
                     const float* __restrict__ ge,
                     const float* __restrict__ oe,
                     const bf16*  __restrict__ W2t,
                     const float* __restrict__ be2,
                     bf16*  __restrict__ edges_out,
                     float* __restrict__ messages,
                     int step0) {
  // A tile (64 x KA_E bf16, 99328B) aliased with f32 h tile (64 x HS, 66560B):
  // A is dead once the layer-1 K loop finishes (barrier-separated).
  __shared__ __attribute__((aligned(16))) unsigned char Ubuf[64 * KA_E * 2];
  __shared__ __attribute__((aligned(16))) bf16  BL[256][BLS];   // B slice 32K x 256N
  __shared__ __attribute__((aligned(16))) bf16  HB[64][HBS];
  __shared__ int recvL[64];

  bf16  (*A)[KA_E] = (bf16 (*)[KA_E])Ubuf;
  float (*H)[HS]   = (float (*)[HS])Ubuf;

  const int tid   = threadIdx.x;
  const int mBase = blockIdx.x * 64;

  // ---- stage A rows: [edge_feat | nodes[sender] | nodes[receiver]] ----
  {
    int row = tid >> 2, sub = tid & 3;        // 4 threads per row
    int m = mBase + row;
    u32x4* dst = (u32x4*)(&A[row][0]);
    const u32x4* s1 = (const u32x4*)(nodes_bf + (size_t)senders[m] * DD);
    const u32x4* s2 = (const u32x4*)(nodes_bf + (size_t)receivers[m] * DD);
    if (step0) {
      if (sub == 0) {
        #pragma unroll
        for (int i = 0; i < 32; ++i) A[row][i] = (bf16)0.0f;
        A[row][0] = (bf16)edges0[m * 3 + 0];
        A[row][1] = (bf16)edges0[m * 3 + 1];
        A[row][2] = (bf16)edges0[m * 3 + 2];
      }
      #pragma unroll
      for (int i = 0; i < 8; ++i) {
        g2l_b128(&dst[4  + sub * 8 + i], &s1[sub * 8 + i]);  // k = 32..287
        g2l_b128(&dst[36 + sub * 8 + i], &s2[sub * 8 + i]);  // k = 288..543
      }
    } else {
      const u32x4* s0 = (const u32x4*)(edges_prev + (size_t)m * DD);
      #pragma unroll
      for (int i = 0; i < 8; ++i) {
        g2l_b128(&dst[     sub * 8 + i], &s0[sub * 8 + i]);  // k = 0..255
        g2l_b128(&dst[32 + sub * 8 + i], &s1[sub * 8 + i]);  // k = 256..511
        g2l_b128(&dst[64 + sub * 8 + i], &s2[sub * 8 + i]);  // k = 512..767
      }
    }
    if (tid < 64) recvL[tid] = receivers[mBase + tid] - NSPA;
    async_fence();
  }

  const int lane = tid & 31, w = tid >> 5;
  const int band = w >> 1, half = w & 1;
  const int arow = band * 16 + (lane & 15);
  const int lh   = lane >> 4;
  const int ncol = lane & 15;

  f32x8 acc[8] = {};

  // ---- layer 1: [64 x kpad] @ [kpad x 256], B slice staged in LDS ----
  const int KB = kpad >> 5;
  for (int kb = 0; kb < KB; ++kb) {
    __syncthreads();                 // A ready (1st iter) / BL consumed
    {                                // stage B slice: thread tid owns col tid
      const u32x4* src = (const u32x4*)(W1t + (size_t)tid * kpad + kb * 32);
      u32x4* d = (u32x4*)(&BL[tid][0]);
      #pragma unroll
      for (int i = 0; i < 4; ++i) g2l_b128(&d[i], &src[i]);
      async_fence();
    }
    __syncthreads();                 // BL ready
    AF a;
    const u32x4* ap = (const u32x4*)(&A[arow][kb * 32 + lh * 8]);
    a.u[0] = ap[0]; a.u[1] = ap[2];
    #pragma unroll
    for (int t = 0; t < 8; ++t) {
      int n = half * 128 + t * 16 + ncol;
      const u32x4* bp = (const u32x4*)(&BL[n][lh * 8]);
      AF b; b.u[0] = bp[0]; b.u[1] = bp[2];
      acc[t] = wmma_bf16(a.v, b.v, acc[t]);
    }
  }
  __syncthreads();                   // everyone done reading A -> reuse as H

  // bias + relu -> H (f32, aliases A)
  #pragma unroll
  for (int t = 0; t < 8; ++t) {
    int n = half * 128 + t * 16 + ncol;
    float bias = be1[n];
    #pragma unroll
    for (int j = 0; j < 8; ++j)
      H[band * 16 + j + 8 * lh][n] = fmaxf(acc[t][j] + bias, 0.0f);
  }
  __syncthreads();

  // ---- layernorm (f32), emit bf16 hhat ----
  {
    int r = w * 8;                   // 8 rows per wave
    for (int rr = 0; rr < 8; ++rr, ++r) {
      float s = 0.0f, s2 = 0.0f;
      #pragma unroll
      for (int c = lane; c < DD; c += 32) { float x = H[r][c]; s += x; s2 += x * x; }
      #pragma unroll
      for (int off = 16; off > 0; off >>= 1) {
        s  += __shfl_xor(s,  off);
        s2 += __shfl_xor(s2, off);
      }
      float mean = s * (1.0f / DD);
      float var  = s2 * (1.0f / DD) - mean * mean;
      float rs   = rsqrtf(var + 1e-5f);
      #pragma unroll
      for (int c = lane; c < DD; c += 32)
        HB[r][c] = (bf16)((H[r][c] - mean) * rs * ge[c] + oe[c]);
    }
  }

  // ---- layer 2: [64 x 256] @ [256 x 256], B slice staged in LDS ----
  f32x8 acc2[8] = {};
  for (int kb = 0; kb < 8; ++kb) {
    __syncthreads();                 // HB ready (1st iter) / BL consumed
    {
      const u32x4* src = (const u32x4*)(W2t + (size_t)tid * DD + kb * 32);
      u32x4* d = (u32x4*)(&BL[tid][0]);
      #pragma unroll
      for (int i = 0; i < 4; ++i) g2l_b128(&d[i], &src[i]);
      async_fence();
    }
    __syncthreads();
    AF a;
    const u32x4* ap = (const u32x4*)(&HB[arow][kb * 32 + lh * 8]);
    a.u[0] = ap[0]; a.u[1] = ap[2];
    #pragma unroll
    for (int t = 0; t < 8; ++t) {
      int n = half * 128 + t * 16 + ncol;
      const u32x4* bp = (const u32x4*)(&BL[n][lh * 8]);
      AF b; b.u[0] = bp[0]; b.u[1] = bp[2];
      acc2[t] = wmma_bf16(a.v, b.v, acc2[t]);
    }
  }

  // ---- epilogue: new edges (bf16) + segment-sum scatter (f32 atomics) ----
  #pragma unroll
  for (int t = 0; t < 8; ++t) {
    int n = half * 128 + t * 16 + ncol;
    float bias = be2[n];
    #pragma unroll
    for (int j = 0; j < 8; ++j) {
      int rl = band * 16 + j + 8 * lh;
      int m  = mBase + rl;
      float val = acc2[t][j] + bias;
      edges_out[(size_t)m * DD + n] = (bf16)val;
      atomicAdd(&messages[(size_t)recvL[rl] * DD + n], val);
    }
  }
}

// ---------------- fused node MLP ----------------
// grid = ceil(NSPH/32), 256 threads. Updates sphere rows of nodes_bf in place
// (each row owned by exactly one block); writes f32 output on the last step.
__global__ __launch_bounds__(256)
void node_mlp_kernel(const float* __restrict__ messages,
                     const bf16*  __restrict__ Wn1t,
                     const float* __restrict__ bn1,
                     const float* __restrict__ gn,
                     const float* __restrict__ on_,
                     const bf16*  __restrict__ Wn2t,
                     const float* __restrict__ bn2,
                     bf16*  __restrict__ nodes_bf,   // read old / write new sphere rows
                     float* __restrict__ out_f32,
                     int writeOut) {
  __shared__ __attribute__((aligned(16))) bf16  A[32][KA_N];
  __shared__ __attribute__((aligned(16))) float H[32][HS];
  __shared__ __attribute__((aligned(16))) bf16  HB[32][HBS];

  const int tid   = threadIdx.x;
  const int mBase = blockIdx.x * 32;

  // ---- stage A rows: [sphere_bf | bf16(messages)] ----
  {
    int row = tid >> 3, sub = tid & 7;
    int m = mBase + row;
    u32x4* dst = (u32x4*)(&A[row][0]);
    if (m < NSPH) {
      const u32x4* s0 = (const u32x4*)(nodes_bf + ((size_t)(NSPA + m)) * DD);
      #pragma unroll
      for (int i = 0; i < 4; ++i) g2l_b128(&dst[sub * 4 + i], &s0[sub * 4 + i]);
      const float* mp = messages + (size_t)m * DD + sub * 32;
      bf16* dp = &A[row][DD + sub * 32];
      #pragma unroll
      for (int i = 0; i < 32; ++i) dp[i] = (bf16)mp[i];
    } else {
      u32x4 z = {0u, 0u, 0u, 0u};
      #pragma unroll
      for (int i = 0; i < 8; ++i) dst[sub * 8 + i] = z;
    }
    async_fence();
  }
  __syncthreads();

  const int lane = tid & 31, w = tid >> 5;
  const int band = w >> 2, q = w & 3;
  const int arow = band * 16 + (lane & 15);
  const int lh   = lane >> 4;
  const int ncol = lane & 15;

  f32x8 acc[4] = {};

  // ---- layer 1: [32 x 512] @ [512 x 256] ----
  #pragma unroll 4
  for (int kb = 0; kb < (KPADN >> 5); ++kb) {
    AF a;
    const u32x4* ap = (const u32x4*)(&A[arow][kb * 32 + lh * 8]);
    a.u[0] = ap[0]; a.u[1] = ap[2];
    #pragma unroll
    for (int t = 0; t < 4; ++t) {
      int n = q * 64 + t * 16 + ncol;
      const u32x4* bp = (const u32x4*)(Wn1t + (size_t)n * KPADN + kb * 32 + lh * 8);
      AF b; b.u[0] = bp[0]; b.u[1] = bp[2];
      acc[t] = wmma_bf16(a.v, b.v, acc[t]);
    }
  }

  #pragma unroll
  for (int t = 0; t < 4; ++t) {
    int n = q * 64 + t * 16 + ncol;
    float bias = bn1[n];
    #pragma unroll
    for (int j = 0; j < 8; ++j)
      H[band * 16 + j + 8 * lh][n] = fmaxf(acc[t][j] + bias, 0.0f);
  }
  __syncthreads();

  {
    int r = w * 4;
    for (int rr = 0; rr < 4; ++rr, ++r) {
      float s = 0.0f, s2 = 0.0f;
      #pragma unroll
      for (int c = lane; c < DD; c += 32) { float x = H[r][c]; s += x; s2 += x * x; }
      #pragma unroll
      for (int off = 16; off > 0; off >>= 1) {
        s  += __shfl_xor(s,  off);
        s2 += __shfl_xor(s2, off);
      }
      float mean = s * (1.0f / DD);
      float var  = s2 * (1.0f / DD) - mean * mean;
      float rs   = rsqrtf(var + 1e-5f);
      #pragma unroll
      for (int c = lane; c < DD; c += 32)
        HB[r][c] = (bf16)((H[r][c] - mean) * rs * gn[c] + on_[c]);
    }
  }
  __syncthreads();

  f32x8 acc2[4] = {};
  #pragma unroll
  for (int kb = 0; kb < 8; ++kb) {
    AF a;
    const u32x4* ap = (const u32x4*)(&HB[arow][kb * 32 + lh * 8]);
    a.u[0] = ap[0]; a.u[1] = ap[2];
    #pragma unroll
    for (int t = 0; t < 4; ++t) {
      int n = q * 64 + t * 16 + ncol;
      const u32x4* bp = (const u32x4*)(Wn2t + (size_t)n * DD + kb * 32 + lh * 8);
      AF b; b.u[0] = bp[0]; b.u[1] = bp[2];
      acc2[t] = wmma_bf16(a.v, b.v, acc2[t]);
    }
  }

  #pragma unroll
  for (int t = 0; t < 4; ++t) {
    int n = q * 64 + t * 16 + ncol;
    float bias = bn2[n];
    #pragma unroll
    for (int j = 0; j < 8; ++j) {
      int rl = band * 16 + j + 8 * lh;
      int m  = mBase + rl;
      if (m < NSPH) {
        float val = acc2[t][j] + bias;
        nodes_bf[((size_t)(NSPA + m)) * DD + n] = (bf16)val;
        if (writeOut) out_f32[(size_t)m * DD + n] = val;
      }
    }
  }
}

// ---------------- launch ----------------

extern "C" void kernel_launch(void* const* d_in, const int* in_sizes, int n_in,
                              void* d_out, int out_size, void* d_ws, size_t ws_size,
                              hipStream_t stream) {
  (void)in_sizes; (void)n_in; (void)out_size; (void)ws_size;
  const float* nodes   = (const float*)d_in[0];
  const float* edges0  = (const float*)d_in[1];
  const int*   senders = (const int*)d_in[2];
  const int*   recvs   = (const int*)d_in[3];
  const float* We1a    = (const float*)d_in[4];
  const float* We1b    = (const float*)d_in[5];
  const float* be1     = (const float*)d_in[6];
  const float* ge      = (const float*)d_in[7];
  const float* oe      = (const float*)d_in[8];
  const float* We2     = (const float*)d_in[9];
  const float* be2     = (const float*)d_in[10];
  const float* Wn1     = (const float*)d_in[11];
  const float* bn1     = (const float*)d_in[12];
  const float* gn      = (const float*)d_in[13];
  const float* on_     = (const float*)d_in[14];
  const float* Wn2     = (const float*)d_in[15];
  const float* bn2     = (const float*)d_in[16];
  float* out = (float*)d_out;

  char* ws = (char*)d_ws;
  size_t off = 0;
  auto alloc = [&](size_t bytes) -> char* {
    char* p = ws + off;
    off = (off + bytes + 255) & ~(size_t)255;
    return p;
  };
  bf16*  nodes_bf = (bf16*) alloc((size_t)(NSPA + NSPH) * DD * sizeof(bf16));
  bf16*  edges_bf = (bf16*) alloc((size_t)NEDGE * DD * sizeof(bf16));
  float* messages = (float*)alloc((size_t)NSPH * DD * sizeof(float));
  bf16*  W1at     = (bf16*) alloc((size_t)DD * KPAD0 * sizeof(bf16));
  bf16*  W1bt     = (bf16*) alloc((size_t)DD * KPAD1 * sizeof(bf16));
  bf16*  We2t     = (bf16*) alloc((size_t)DD * DD * sizeof(bf16));
  bf16*  Wn1t     = (bf16*) alloc((size_t)DD * KPADN * sizeof(bf16));
  bf16*  Wn2t     = (bf16*) alloc((size_t)DD * DD * sizeof(bf16));

  // one-time (per launch) bf16 prep
  {
    int n = (NSPA + NSPH) * DD;
    cvt_bf16_kernel<<<(n + 255) / 256, 256, 0, stream>>>(nodes, nodes_bf, n);
    prep_w1a_kernel<<<(256 * KPAD0 + 255) / 256, 256, 0, stream>>>(We1a, W1at);
    prep_wt_kernel<<<(256 * KPAD1 + 255) / 256, 256, 0, stream>>>(We1b, W1bt, KPAD1);
    prep_wt_kernel<<<(256 * DD    + 255) / 256, 256, 0, stream>>>(We2,  We2t, DD);
    prep_wt_kernel<<<(256 * KPADN + 255) / 256, 256, 0, stream>>>(Wn1,  Wn1t, KPADN);
    prep_wt_kernel<<<(256 * DD    + 255) / 256, 256, 0, stream>>>(Wn2,  Wn2t, DD);
  }

  const int edgeBlocks = NEDGE / 64;            // 2048
  const int nodeBlocks = (NSPH + 31) / 32;      // 91
  const int msgElems   = NSPH * DD;

  for (int step = 0; step < 3; ++step) {
    zero_f32_kernel<<<(msgElems + 255) / 256, 256, 0, stream>>>(messages, msgElems);
    edge_mlp_kernel<<<edgeBlocks, 256, 0, stream>>>(
        edges0, edges_bf, nodes_bf, senders, recvs,
        (step == 0) ? W1at : W1bt, (step == 0) ? KPAD0 : KPAD1,
        be1, ge, oe, We2t, be2, edges_bf, messages, (step == 0) ? 1 : 0);
    node_mlp_kernel<<<nodeBlocks, 256, 0, stream>>>(
        messages, Wn1t, bn1, gn, on_, Wn2t, bn2,
        nodes_bf, out, (step == 2) ? 1 : 0);
  }
}
